// Planner_43611097923933
// MI455X (gfx1250) — matrix operand order repaired and checked by reading, not compiled, not run on hardware
//
#include <hip/hip_runtime.h>

// ---------------------------------------------------------------------------
// xLSTM planner (B=4, T=256, D=512, L=2, H=8, I=1024, DH=128) + MDN head
// CDNA5 / gfx1250: bf16 V_WMMA_F32_16X16X32_BF16 for all weight GEMMs,
// persistent kernel with global barrier across 32 workgroups.
// ---------------------------------------------------------------------------

#define NWG 32
#define BT  128

typedef __attribute__((ext_vector_type(16))) __bf16 bf16x16;
typedef __attribute__((ext_vector_type(8)))  float  f32x8;

union ABu { bf16x16 v; unsigned u[8]; uint4 q[2]; };

__device__ __forceinline__ unsigned short f2bf(float f) {
  unsigned u = __float_as_uint(f);
  unsigned r = (u + 0x7fffu + ((u >> 16) & 1u)) >> 16;   // round-to-nearest-even
  return (unsigned short)r;
}

struct Params {
  const float *x, *ln_s, *ln_b, *Wi, *bi, *Wf, *bf, *gn_s, *gn_b;
  const float *fln_s, *fln_b, *b_pi, *b_mu, *b_sig;
  const unsigned short *swzProj, *swzQ, *swzK, *swzV, *swzDn, *swzPi, *swzMu, *swzSg;
  float *C, *nst, *mst, *h, *xmf, *z, *qb, *kb, *vb, *hp, *out;
  unsigned short *xmb, *hhb;
  unsigned *gbar;
};

// global sense-reversing barrier (all NWG workgroups resident)
__device__ __forceinline__ void gbarrier(unsigned *g) {
  __syncthreads();
  if (threadIdx.x == 0) {
    __threadfence();
    unsigned gen = atomicAdd(&g[1], 0u);
    unsigned arr = atomicAdd(&g[0], 1u);
    if (arr == NWG - 1u) {
      atomicExch(&g[0], 0u);
      __threadfence();
      atomicAdd(&g[1], 1u);
    } else {
      while (atomicAdd(&g[1], 0u) == gen) __builtin_amdgcn_s_sleep(1);
    }
    __threadfence();
  }
  __syncthreads();
}

// 16x16 output tile: A (16 x 32*ktiles bf16, LDS, WMMA A layout rows) x
// swizzled bf16 weights, K-loop of V_WMMA_F32_16X16X32_BF16
__device__ __forceinline__ f32x8 gemm_tile(const unsigned short *__restrict__ smA,
                                           int Krow, int ktiles,
                                           const unsigned short *__restrict__ wbase,
                                           int ntilesN, int nt, int lane) {
  f32x8 acc = {0.f, 0.f, 0.f, 0.f, 0.f, 0.f, 0.f, 0.f};
  const int row = lane & 15;
  const int lh  = (lane >> 4) << 3;        // +8 K offset for lanes 16..31
  for (int kt = 0; kt < ktiles; ++kt) {
    ABu a, b;
    const unsigned short *ar = smA + row * Krow + kt * 32;
#pragma unroll
    for (int e = 0; e < 8; ++e) {
      int ko = 2 * e + ((e >= 4) ? 8 : 0) + lh;  // K pairs {0..7,16..23} / {8..15,24..31}
      a.u[e] = *(const unsigned *)(ar + ko);
    }
    const unsigned short *wB = wbase + (((size_t)(kt * ntilesN + nt)) << 9) + lane * 16;
    b.q[0] = ((const uint4 *)wB)[0];
    b.q[1] = ((const uint4 *)wB)[1];
    if (kt + 1 < ktiles)
      __builtin_prefetch(wbase + (((size_t)((kt + 1) * ntilesN + nt)) << 9) + lane * 16, 0, 0);
    acc = __builtin_amdgcn_wmma_f32_16x16x32_bf16(false, a.v, false, b.v,
                                                  (short)0, acc, false, false);
  }
  return acc;
}

// f32 (layers,K,N) row-major -> bf16 swizzled WMMA-B layout
__global__ void swz_kernel(const float *__restrict__ src, unsigned short *__restrict__ dst,
                           int Kdim, int Ndim, int layers) {
  size_t total = (size_t)layers * Kdim * Ndim;
  size_t stride = (size_t)gridDim.x * blockDim.x;
  for (size_t idx = (size_t)blockIdx.x * blockDim.x + threadIdx.x; idx < total; idx += stride) {
    size_t per = (size_t)Kdim * Ndim;
    int l = (int)(idx / per);
    size_t r = idx % per;
    int k = (int)(r / Ndim), n = (int)(r % Ndim);
    int kt = k >> 5, kr = k & 31, nt = n >> 4, nr = n & 15;
    int lane = nr + ((kr & 8) << 1);
    int j = (kr & 7) | ((kr & 16) >> 1);
    dst[(size_t)l * per + (((size_t)(kt * (Ndim >> 4) + nt)) << 9) + lane * 16 + j] =
        f2bf(src[idx]);
  }
}

__global__ void zero_kernel(unsigned *__restrict__ p, size_t nwords) {
  size_t stride = (size_t)gridDim.x * blockDim.x;
  for (size_t i = (size_t)blockIdx.x * blockDim.x + threadIdx.x; i < nwords; i += stride)
    p[i] = 0u;
}

__global__ __launch_bounds__(BT, 1) void xlstm_kernel(Params p) {
  __shared__ unsigned short smA[16 * 1024];  // WMMA A staging (bf16)
  __shared__ float sh[2048];                 // h staging / q,k,v slices
  __shared__ float red[BT];
  __shared__ float scal[16];
  __shared__ float ptile[4 * 4 * 16];

  const int tid  = threadIdx.x;
  const int wg   = blockIdx.x;
  const int wv   = tid >> 5;
  const int lane = tid & 31;
  const int gwave = wg * 4 + wv;
  const float inv_sqrt_dh = 0.08838834764831845f;  // 1/sqrt(128)

  for (int t = 0; t < 256; ++t) {
    // ---- Phase S: h = x[:, t, :] ----
    {
      int gid = wg * BT + tid;
      if (gid < 2048) {
        int b = gid >> 9, d = gid & 511;
        p.h[gid] = p.x[((size_t)b * 256 + t) * 512 + d];
      }
    }
    gbarrier(p.gbar);

    for (int l = 0; l < 2; ++l) {
      // ---- P0: LayerNorm(h) -> smA bf16 [16][512] (redundant per WG) ----
      for (int i = tid; i < 2048; i += BT) sh[i] = p.h[i];
      __syncthreads();
      {
        int b = tid >> 5, sub = tid & 31;
        float s = 0.f, s2 = 0.f;
        for (int d = sub; d < 512; d += 32) { float v = sh[b * 512 + d]; s += v; s2 += v * v; }
        red[tid] = s; __syncthreads();
        if (tid < 4) { float m = 0; for (int j = 0; j < 32; ++j) m += red[tid * 32 + j];
                       scal[tid] = m * (1.f / 512.f); }
        __syncthreads();
        red[tid] = s2; __syncthreads();
        if (tid < 4) { float q = 0; for (int j = 0; j < 32; ++j) q += red[tid * 32 + j];
                       float mn = scal[tid];
                       scal[4 + tid] = rsqrtf(q * (1.f / 512.f) - mn * mn + 1e-5f); }
        __syncthreads();
      }
      for (int i = tid; i < 2048; i += BT) {
        int b = i >> 9, d = i & 511;
        float v = (sh[i] - scal[b]) * scal[4 + b] * p.ln_s[l * 512 + d] + p.ln_b[l * 512 + d];
        smA[b * 512 + d] = f2bf(v);
      }
      for (int i = 2048 + tid; i < 8192; i += BT) smA[i] = 0;  // pad rows 4..15
      __syncthreads();

      // ---- P1: xp = xn @ W_proj  (2048 cols, 1 n-tile per wave) ----
      {
        int nt = gwave;  // 0..127
        f32x8 acc = gemm_tile(smA, 512, 16, p.swzProj + (size_t)l * 512 * 2048, 128, nt, lane);
        if (lane < 16) {
#pragma unroll
          for (int r = 0; r < 4; ++r) {
            int col = nt * 16 + lane;
            float v = acc[r];
            if (col < 1024) { p.xmf[r * 1024 + col] = v; p.xmb[r * 1024 + col] = f2bf(v); }
            else            { p.z[r * 1024 + (col - 1024)] = v; }
          }
        }
      }
      gbarrier(p.gbar);

      // ---- P2: q,k,v = xm @ Wq/Wk/Wv ----
      {
        for (int i = tid; i < 8192; i += BT)
          ((unsigned *)smA)[i] = ((const unsigned *)p.xmb)[i];
        __syncthreads();
        for (int tg = gwave; tg < 192; tg += 128) {
          int mat = tg >> 6, nt = tg & 63;
          const unsigned short *wsrc =
              (mat == 0 ? p.swzQ : mat == 1 ? p.swzK : p.swzV) + (size_t)l * 1024 * 1024;
          f32x8 acc = gemm_tile(smA, 1024, 32, wsrc, 64, nt, lane);
          float scl = (mat == 1) ? inv_sqrt_dh : 1.f;
          float *dst = (mat == 0 ? p.qb : mat == 1 ? p.kb : p.vb);
          if (lane < 16) {
#pragma unroll
            for (int r = 0; r < 4; ++r) dst[r * 1024 + nt * 16 + lane] = acc[r] * scl;
          }
        }
      }
      gbarrier(p.gbar);

      // ---- P3: per-(b,head) gates + matrix memory update (one WG each) ----
      {
        int b = wg >> 3, hd = wg & 7;
        float pi_ = 0.f, pf_ = 0.f;
        for (int k = tid; k < 1024; k += BT) {
          float xv = p.xmf[b * 1024 + k];
          pi_ += xv * p.Wi[((size_t)l * 1024 + k) * 8 + hd];
          pf_ += xv * p.Wf[((size_t)l * 1024 + k) * 8 + hd];
        }
        red[tid] = pi_; __syncthreads();
        if (tid == 0) { float s = 0; for (int j = 0; j < BT; ++j) s += red[j];
                        scal[0] = s + p.bi[l * 8 + hd]; }
        __syncthreads();
        red[tid] = pf_; __syncthreads();
        if (tid == 0) {
          float s = 0; for (int j = 0; j < BT; ++j) s += red[j];
          float fp = s + p.bf[l * 8 + hd];
          float lf = (fp >= 0.f) ? -log1pf(expf(-fp)) : (fp - log1pf(expf(fp)));
          int mi = (l * 4 + b) * 8 + hd;
          float mo = p.mst[mi];
          float ipre = scal[0];
          float mn = fmaxf(lf + mo, ipre);
          scal[1] = expf(ipre - mn);       // ig
          scal[2] = expf(lf + mo - mn);    // fg
          p.mst[mi] = mn;
        }
        __syncthreads();
        float ig = scal[1], fg = scal[2];
        sh[tid]       = p.qb[b * 1024 + hd * 128 + tid];
        sh[128 + tid] = p.kb[b * 1024 + hd * 128 + tid];
        sh[256 + tid] = p.vb[b * 1024 + hd * 128 + tid];
        __syncthreads();
        float *Cb = p.C + ((size_t)((l * 4 + b) * 8 + hd)) * 16384;
        int vc = tid;
        float vvv = sh[256 + vc];
        float acc = 0.f;
        for (int kk = 0; kk < 128; ++kk) {
          float cv = fg * Cb[kk * 128 + vc] + ig * sh[128 + kk] * vvv;
          Cb[kk * 128 + vc] = cv;
          acc += sh[kk] * cv;
        }
        float *nb = p.nst + ((size_t)((l * 4 + b) * 8 + hd)) * 128;
        float nn = fg * nb[vc] + ig * sh[128 + vc];
        nb[vc] = nn;
        red[tid] = sh[vc] * nn; __syncthreads();
        if (tid == 0) { float s = 0; for (int j = 0; j < BT; ++j) s += red[j];
                        scal[3] = fmaxf(fabsf(s), 1.f); }
        __syncthreads();
        float hh = acc / scal[3];
        red[tid] = hh; __syncthreads();
        if (tid == 0) { float s = 0; for (int j = 0; j < BT; ++j) s += red[j];
                        scal[4] = s * (1.f / 128.f); }
        __syncthreads();
        float c0 = hh - scal[4];
        red[tid] = c0 * c0; __syncthreads();
        if (tid == 0) { float s = 0; for (int j = 0; j < BT; ++j) s += red[j];
                        scal[5] = rsqrtf(s * (1.f / 128.f) + 1e-5f); }
        __syncthreads();
        int col = hd * 128 + vc;
        float val = c0 * scal[5] * p.gn_s[l * 1024 + col] + p.gn_b[l * 1024 + col];
        float zv = p.z[b * 1024 + col];
        val *= zv / (1.f + expf(-zv));     // * silu(z)
        p.hhb[b * 1024 + col] = f2bf(val);
      }
      gbarrier(p.gbar);

      // ---- P4: h = skip + hh @ W_down (K split across 4 waves) ----
      {
        for (int i = tid; i < 8192; i += BT)
          ((unsigned *)smA)[i] = ((const unsigned *)p.hhb)[i];
        __syncthreads();
        int nt = wg;  // 0..31 -> 512 cols
        f32x8 acc = {0.f, 0.f, 0.f, 0.f, 0.f, 0.f, 0.f, 0.f};
        {
          const unsigned short *wbase = p.swzDn + (size_t)l * 1024 * 512;
          int row = lane & 15, lh = (lane >> 4) << 3;
          for (int kt = wv * 8; kt < wv * 8 + 8; ++kt) {
            ABu a, b;
            const unsigned short *ar = smA + row * 1024 + kt * 32;
#pragma unroll
            for (int e = 0; e < 8; ++e) {
              int ko = 2 * e + ((e >= 4) ? 8 : 0) + lh;
              a.u[e] = *(const unsigned *)(ar + ko);
            }
            const unsigned short *wB = wbase + (((size_t)(kt * 32 + nt)) << 9) + lane * 16;
            b.q[0] = ((const uint4 *)wB)[0];
            b.q[1] = ((const uint4 *)wB)[1];
            acc = __builtin_amdgcn_wmma_f32_16x16x32_bf16(false, a.v, false, b.v,
                                                          (short)0, acc, false, false);
          }
        }
        if (lane < 16) {
#pragma unroll
          for (int r = 0; r < 4; ++r) ptile[(wv * 4 + r) * 16 + lane] = acc[r];
        }
        __syncthreads();
        if (tid < 64) {
          int r = tid >> 4, n = tid & 15;
          float s = ptile[r * 16 + n] + ptile[(4 + r) * 16 + n] +
                    ptile[(8 + r) * 16 + n] + ptile[(12 + r) * 16 + n];
          p.h[r * 512 + wg * 16 + n] += s;
        }
      }
      gbarrier(p.gbar);
    }  // layers
  }    // t

  // ---- Final LN + MDN heads ----
  {
    for (int i = tid; i < 2048; i += BT) sh[i] = p.h[i];
    __syncthreads();
    {
      int b = tid >> 5, sub = tid & 31;
      float s = 0.f, s2 = 0.f;
      for (int d = sub; d < 512; d += 32) { float v = sh[b * 512 + d]; s += v; s2 += v * v; }
      red[tid] = s; __syncthreads();
      if (tid < 4) { float m = 0; for (int j = 0; j < 32; ++j) m += red[tid * 32 + j];
                     scal[tid] = m * (1.f / 512.f); }
      __syncthreads();
      red[tid] = s2; __syncthreads();
      if (tid < 4) { float q = 0; for (int j = 0; j < 32; ++j) q += red[tid * 32 + j];
                     float mn = scal[tid];
                     scal[4 + tid] = rsqrtf(q * (1.f / 512.f) - mn * mn + 1e-5f); }
      __syncthreads();
    }
    for (int i = tid; i < 2048; i += BT) {
      int b = i >> 9, d = i & 511;
      float v = (sh[i] - scal[b]) * scal[4 + b] * p.fln_s[d] + p.fln_b[d];
      smA[b * 512 + d] = f2bf(v);
    }
    for (int i = 2048 + tid; i < 8192; i += BT) smA[i] = 0;
    __syncthreads();
    for (int tg = gwave; tg < 258; tg += 128) {
      int nt, ntilesN, coff;
      const unsigned short *wsrc; const float *bias;
      if (tg < 2)        { nt = tg;       ntilesN = 2;   wsrc = p.swzPi; bias = p.b_pi;  coff = 0; }
      else if (tg < 130) { nt = tg - 2;   ntilesN = 128; wsrc = p.swzMu; bias = p.b_mu;  coff = 32; }
      else               { nt = tg - 130; ntilesN = 128; wsrc = p.swzSg; bias = p.b_sig; coff = 2080; }
      f32x8 acc = gemm_tile(smA, 512, 16, wsrc, ntilesN, nt, lane);
      if (lane < 16) {
#pragma unroll
        for (int r = 0; r < 4; ++r) {
          int cl = nt * 16 + lane;
          p.hp[r * 4128 + coff + cl] = acc[r] + bias[cl];
        }
      }
    }
  }
  gbarrier(p.gbar);

  // ---- Finalize: softmax(pi), mu, softplus(sigma) -> d_out ----
  {
    int gid = wg * BT + tid;
    if (gid < 4) {
      int b = gid;
      float mx = -1e30f;
      for (int j = 0; j < 32; ++j) mx = fmaxf(mx, p.hp[b * 4128 + j]);
      float s = 0.f;
      for (int j = 0; j < 32; ++j) s += expf(p.hp[b * 4128 + j] - mx);
      float inv = 1.f / s;
      for (int j = 0; j < 32; ++j) p.out[b * 32 + j] = expf(p.hp[b * 4128 + j] - mx) * inv;
    }
    for (int e = gid; e < 8192; e += NWG * BT) {
      int b = e >> 11, kg = e & 2047;
      p.out[128 + e] = p.hp[b * 4128 + 32 + kg];
      float v = p.hp[b * 4128 + 2080 + kg];
      p.out[8320 + e] = (v > 20.f) ? v : log1pf(expf(v));
    }
  }
}

extern "C" void kernel_launch(void *const *d_in, const int *in_sizes, int n_in,
                              void *d_out, int out_size, void *d_ws, size_t ws_size,
                              hipStream_t stream) {
  (void)in_sizes; (void)n_in; (void)out_size; (void)ws_size;
  const float *x     = (const float *)d_in[0];
  const float *ln_s  = (const float *)d_in[1];
  const float *ln_b  = (const float *)d_in[2];
  const float *Wproj = (const float *)d_in[3];
  const float *Wq    = (const float *)d_in[4];
  const float *Wk    = (const float *)d_in[5];
  const float *Wv    = (const float *)d_in[6];
  const float *Wi    = (const float *)d_in[7];
  const float *bi    = (const float *)d_in[8];
  const float *Wf    = (const float *)d_in[9];
  const float *bfg   = (const float *)d_in[10];
  const float *gn_s  = (const float *)d_in[11];
  const float *gn_b  = (const float *)d_in[12];
  const float *Wdown = (const float *)d_in[13];
  const float *fln_s = (const float *)d_in[14];
  const float *fln_b = (const float *)d_in[15];
  const float *W_pi  = (const float *)d_in[16];
  const float *b_pi  = (const float *)d_in[17];
  const float *W_mu  = (const float *)d_in[18];
  const float *b_mu  = (const float *)d_in[19];
  const float *W_sig = (const float *)d_in[20];
  const float *b_sig = (const float *)d_in[21];

  char *ws = (char *)d_ws;
  size_t o = 0;
  auto al = [&](size_t bytes) { size_t r = o; o += (bytes + 255) & ~(size_t)255; return r; };

  size_t oProj = al((size_t)2 * 512 * 2048 * 2);
  size_t oQ    = al((size_t)2 * 1024 * 1024 * 2);
  size_t oK    = al((size_t)2 * 1024 * 1024 * 2);
  size_t oV    = al((size_t)2 * 1024 * 1024 * 2);
  size_t oDn   = al((size_t)2 * 1024 * 512 * 2);
  size_t oPi   = al((size_t)512 * 32 * 2);
  size_t oMu   = al((size_t)512 * 2048 * 2);
  size_t oSg   = al((size_t)512 * 2048 * 2);
  size_t oZero = o;                           // ---- zero-initialized region ----
  size_t oC    = al((size_t)2 * 4 * 8 * 128 * 128 * 4);
  size_t oNs   = al((size_t)2 * 4 * 8 * 128 * 4);
  size_t oMs   = al(256);
  size_t oXmb  = al((size_t)16 * 1024 * 2);
  size_t oHhb  = al((size_t)16 * 1024 * 2);
  size_t oBar  = al(256);
  size_t zeroWords = (o - oZero) / 4;         // ---- end zero region ----
  size_t oH    = al((size_t)2048 * 4);
  size_t oXmf  = al((size_t)4096 * 4);
  size_t oZ    = al((size_t)4096 * 4);
  size_t oQb   = al((size_t)4096 * 4);
  size_t oKb   = al((size_t)4096 * 4);
  size_t oVb   = al((size_t)4096 * 4);
  size_t oHp   = al((size_t)4 * 4128 * 4);

  // weight swizzle (f32 -> bf16 WMMA-B layout)
  swz_kernel<<<1024, 256, 0, stream>>>(Wproj, (unsigned short *)(ws + oProj), 512, 2048, 2);
  swz_kernel<<<1024, 256, 0, stream>>>(Wq,    (unsigned short *)(ws + oQ),    1024, 1024, 2);
  swz_kernel<<<1024, 256, 0, stream>>>(Wk,    (unsigned short *)(ws + oK),    1024, 1024, 2);
  swz_kernel<<<1024, 256, 0, stream>>>(Wv,    (unsigned short *)(ws + oV),    1024, 1024, 2);
  swz_kernel<<<1024, 256, 0, stream>>>(Wdown, (unsigned short *)(ws + oDn),   1024, 512,  2);
  swz_kernel<<<64,   256, 0, stream>>>(W_pi,  (unsigned short *)(ws + oPi),   512,  32,   1);
  swz_kernel<<<1024, 256, 0, stream>>>(W_mu,  (unsigned short *)(ws + oMu),   512,  2048, 1);
  swz_kernel<<<1024, 256, 0, stream>>>(W_sig, (unsigned short *)(ws + oSg),   512,  2048, 1);
  // zero state (C, n, m), padded bf16 activation rows, barrier words
  zero_kernel<<<512, 256, 0, stream>>>((unsigned *)(ws + oZero), zeroWords);

  Params p;
  p.x = x; p.ln_s = ln_s; p.ln_b = ln_b; p.Wi = Wi; p.bi = bi; p.Wf = Wf; p.bf = bfg;
  p.gn_s = gn_s; p.gn_b = gn_b; p.fln_s = fln_s; p.fln_b = fln_b;
  p.b_pi = b_pi; p.b_mu = b_mu; p.b_sig = b_sig;
  p.swzProj = (const unsigned short *)(ws + oProj);
  p.swzQ    = (const unsigned short *)(ws + oQ);
  p.swzK    = (const unsigned short *)(ws + oK);
  p.swzV    = (const unsigned short *)(ws + oV);
  p.swzDn   = (const unsigned short *)(ws + oDn);
  p.swzPi   = (const unsigned short *)(ws + oPi);
  p.swzMu   = (const unsigned short *)(ws + oMu);
  p.swzSg   = (const unsigned short *)(ws + oSg);
  p.C   = (float *)(ws + oC);
  p.nst = (float *)(ws + oNs);
  p.mst = (float *)(ws + oMs);
  p.h   = (float *)(ws + oH);
  p.xmf = (float *)(ws + oXmf);
  p.z   = (float *)(ws + oZ);
  p.qb  = (float *)(ws + oQb);
  p.kb  = (float *)(ws + oKb);
  p.vb  = (float *)(ws + oVb);
  p.hp  = (float *)(ws + oHp);
  p.out = (float *)d_out;
  p.xmb = (unsigned short *)(ws + oXmb);
  p.hhb = (unsigned short *)(ws + oHhb);
  p.gbar = (unsigned *)(ws + oBar);

  xlstm_kernel<<<NWG, BT, 0, stream>>>(p);
}
